// CLAM_SB_64269890617619
// MI455X (gfx1250) — matrix operand order repaired and checked by reading, not compiled, not run on hardware
//
#include <hip/hip_runtime.h>
#include <hip/hip_fp16.h>

typedef _Float16 h16;
typedef __attribute__((ext_vector_type(16))) _Float16 v16h;
typedef __attribute__((ext_vector_type(8)))  float    v8f;
typedef __attribute__((ext_vector_type(4))) unsigned int v4u;
typedef __attribute__((ext_vector_type(8))) int v8i_t;
typedef __attribute__((ext_vector_type(4))) int v4i_t;

#define N_ROWS 100000
#define NBLK128 782          // ceil(100000/128)
#define NBLK1K  98           // ceil(100000/1024)
#define NCAND   (NBLK1K*64)  // 6272

#if __has_builtin(__builtin_amdgcn_tensor_load_to_lds)
#define HAVE_TDM 1
#else
#define HAVE_TDM 0
#endif

union V16HU { uint4 u[2]; v16h v; };

// A-matrix 16x32 f16 fragment from LDS (row-major, stride in halves).
// lane l: row = l&15, half = l>>4; K runs [half*8, half*8+8) and [16+half*8, ...).
__device__ inline v16h load_a_frag(const h16* lds, int stride, int row0, int k0) {
  int lane = threadIdx.x & 31;
  int r = lane & 15, hf = lane >> 4;
  const h16* p = lds + (row0 + r) * stride + k0 + hf * 8;
  V16HU U;
  U.u[0] = *(const uint4*)(p);
  U.u[1] = *(const uint4*)(p + 16);
  return U.v;
}

// B-matrix 32x16 f16 fragment from LDS (row-major [K][N], stride in halves).
// lane l: col = l&15, half = l>>4; K runs [half*8, half*8+8) and [16+half*8, ...).
__device__ inline v16h load_b_frag(const h16* lds, int stride, int k0, int c0) {
  int lane = threadIdx.x & 31;
  int n = lane & 15, hf = lane >> 4;
  const h16* p = lds + (k0 + hf * 8) * stride + c0 + n;
  v16h v;
#pragma unroll
  for (int s = 0; s < 8; ++s) {
    v[s] = p[s * stride];
    v[s + 8] = p[(16 + s) * stride];
  }
  return v;
}

#if HAVE_TDM
// 2-D TDM tile load, f16 elements (data_size=1 -> 2 bytes).
// Copies tile_h rows x tile_w elems from global (row stride row_stride_elems)
// into LDS at lds_addr (rows packed contiguously, tile_w elems apart).
// Rows/cols beyond tens_h/tens_w are DMA-zeroed in LDS (OOB reads return 0).
__device__ inline void tdm_load_2d(unsigned lds_addr, const void* gptr,
                                   unsigned tile_w, unsigned tile_h,
                                   unsigned tens_w, unsigned tens_h,
                                   unsigned row_stride_elems) {
  unsigned long long ga = (unsigned long long)(size_t)gptr;
  v4u g0;
  g0[0] = 1u;                                   // count=1 valid descriptor
  g0[1] = lds_addr;                             // LDS byte address
  g0[2] = (unsigned)ga;                         // global addr [31:0]
  g0[3] = (unsigned)((ga >> 32) & 0x1ffffffu) | (2u << 30);  // addr[56:32] | type=2
  v8i_t g1;
  g1[0] = 1 << 16;                              // data_size=1 (2B), no multicast/pad
  g1[1] = (int)((tens_w & 0xffffu) << 16);      // tensor_dim0[15:0] @ [63:48]
  g1[2] = (int)(((tens_w >> 16) & 0xffffu) | ((tens_h & 0xffffu) << 16));
  g1[3] = (int)(((tens_h >> 16) & 0xffffu) | ((tile_w & 0xffffu) << 16));
  g1[4] = (int)(tile_h & 0xffffu);              // tile_dim1 (tile_dim2 = 0)
  g1[5] = (int)row_stride_elems;                // tensor_dim0_stride[31:0]
  g1[6] = 0;
  g1[7] = 0;
  v4i_t gz;
  gz[0] = 0; gz[1] = 0; gz[2] = 0; gz[3] = 0;
  v8i_t g4z;
  g4z[0] = 0; g4z[1] = 0; g4z[2] = 0; g4z[3] = 0;
  g4z[4] = 0; g4z[5] = 0; g4z[6] = 0; g4z[7] = 0;
  __builtin_amdgcn_tensor_load_to_lds(g0, g1, gz, gz, g4z, 0);
}
#endif

// ---------------- K0: fp32 -> fp16 weight conversion ----------------
__global__ __launch_bounds__(256) void k_cvt(const float* __restrict__ W1,
                                             const float* __restrict__ Wa,
                                             const float* __restrict__ Wb,
                                             h16* __restrict__ W1h,
                                             h16* __restrict__ Wah,
                                             h16* __restrict__ Wbh) {
  int i = blockIdx.x * 256 + threadIdx.x;
  if (i < 1024 * 512) W1h[i] = (h16)W1[i];
  if (i < 512 * 256) { Wah[i] = (h16)Wa[i]; Wbh[i] = (h16)Wb[i]; }
}

// ---------------- K1: h1 = relu(h @ W1 + b1), f16 out ----------------
__global__ __launch_bounds__(256) void k_fc1(const float* __restrict__ h,
                                             const h16* __restrict__ W1h,
                                             const float* __restrict__ b1,
                                             h16* __restrict__ h1h) {
  __shared__ h16 Ah[128 * 1024];    // 256 KB: full k-extent of the row tile
  __shared__ h16 Bh[2][32 * 128];   // 16 KB: double-buffered W1 chunk
  const int tid = threadIdx.x;
  const int wave = tid >> 5, lane = tid & 31;
  const int row0 = blockIdx.x * 128;

  // stage h tile -> f16 LDS (needs fp32->fp16 conversion, so VGPR path)
  for (int i = tid; i < 128 * 1024 / 4; i += 256) {
    int f = i * 4;
    int r = f >> 10, c = f & 1023;
    float4 v = make_float4(0.f, 0.f, 0.f, 0.f);
    if (row0 + r < N_ROWS) v = *(const float4*)(h + (size_t)(row0 + r) * 1024 + c);
    h16* d = &Ah[r * 1024 + c];
    d[0] = (h16)v.x; d[1] = (h16)v.y; d[2] = (h16)v.z; d[3] = (h16)v.w;
  }

#if HAVE_TDM
  if (tid < 32)   // wave 0 drives the TDM pipeline
    tdm_load_2d((unsigned)(size_t)&Bh[0][0], W1h, 128, 32, 128, 32, 512);
#endif

  const int hf = lane >> 4, nn = lane & 15;
  for (int cg = 0; cg < 4; ++cg) {
    v8f acc[8];
    for (int j = 0; j < 8; ++j)
      for (int e = 0; e < 8; ++e) acc[j][e] = 0.f;

    for (int ks = 0; ks < 32; ++ks) {
      int s = cg * 32 + ks;
#if HAVE_TDM
      if (tid < 32) __builtin_amdgcn_s_wait_tensorcnt(0);
      __syncthreads();
      if (tid < 32 && s + 1 < 128) {
        int ns = s + 1, ncg = ns >> 5, nks = ns & 31;
        tdm_load_2d((unsigned)(size_t)&Bh[ns & 1][0],
                    W1h + (size_t)(nks * 32) * 512 + ncg * 128,
                    128, 32, 128, 32, 512);
      }
      const h16* Bp = &Bh[s & 1][0];
#else
      { // fallback: VGPR staging
        int r = tid >> 3;
        int cpart = (tid & 7) * 16;
        const h16* src = W1h + (size_t)(ks * 32 + r) * 512 + cg * 128 + cpart;
        uint4 a = *(const uint4*)(src);
        uint4 b = *(const uint4*)(src + 8);
        __syncthreads();
        *(uint4*)&Bh[0][r * 128 + cpart]     = a;
        *(uint4*)&Bh[0][r * 128 + cpart + 8] = b;
        __syncthreads();
      }
      const h16* Bp = &Bh[0][0];
#endif
      v16h afrag = load_a_frag(Ah, 1024, wave * 16, ks * 32);
#pragma unroll
      for (int j = 0; j < 8; ++j) {
        v16h bfrag = load_b_frag(Bp, 128, 0, j * 16);
        acc[j] = __builtin_amdgcn_wmma_f32_16x16x32_f16(
            false, afrag, false, bfrag, (short)0, acc[j], false, false);
      }
    }
    // epilogue: bias + relu + f16 store
    for (int j = 0; j < 8; ++j) {
      int col = cg * 128 + j * 16 + nn;
      float bias = b1[col];
#pragma unroll
      for (int i = 0; i < 8; ++i) {
        int grow = row0 + wave * 16 + i + 8 * hf;
        if (grow < N_ROWS) {
          float v = acc[j][i] + bias;
          v = v > 0.f ? v : 0.f;
          h1h[(size_t)grow * 512 + col] = (h16)v;
        }
      }
    }
  }
}

// ------- K2: A_raw = (tanh(h1@Wa+ba) * sigmoid(h1@Wb+bb)) @ Wc + bc -------
__global__ __launch_bounds__(256) void k_attn(const h16* __restrict__ h1h,
                                              const h16* __restrict__ Wah,
                                              const h16* __restrict__ Wbh,
                                              const float* __restrict__ ba,
                                              const float* __restrict__ bb,
                                              const float* __restrict__ Wc,
                                              const float* __restrict__ bc,
                                              float* __restrict__ Araw) {
  __shared__ h16 Hs[128 * 512];     // 128 KB
  __shared__ h16 Ba[2][512 * 16];   // 32 KB double-buffered
  __shared__ h16 Bb[2][512 * 16];   // 32 KB double-buffered
  const int tid = threadIdx.x, wave = tid >> 5, lane = tid & 31;
  const int row0 = blockIdx.x * 128;
  const int hf = lane >> 4, nn = lane & 15;
  const int rows_rem = N_ROWS - row0;   // tail rows get DMA-zeroed via tensor_dim1

#if HAVE_TDM
  if (tid < 32) {
    tdm_load_2d((unsigned)(size_t)&Hs[0], h1h + (size_t)row0 * 512,
                512, 128, 512, (unsigned)rows_rem, 512);
    tdm_load_2d((unsigned)(size_t)&Ba[0][0], Wah, 16, 512, 16, 512, 256);
    tdm_load_2d((unsigned)(size_t)&Bb[0][0], Wbh, 16, 512, 16, 512, 256);
  }
#else
  for (int i = tid; i < 128 * 512 / 8; i += 256) {
    int f = i * 8; int r = f >> 9, c = f & 511;
    uint4 v = make_uint4(0u, 0u, 0u, 0u);
    if (row0 + r < N_ROWS) v = *(const uint4*)(h1h + (size_t)(row0 + r) * 512 + c);
    *(uint4*)&Hs[r * 512 + c] = v;
  }
#endif

  float rp[8];
  for (int i = 0; i < 8; ++i) rp[i] = 0.f;

  for (int j = 0; j < 16; ++j) {   // 16 column tiles of D=256
#if HAVE_TDM
    if (tid < 32) __builtin_amdgcn_s_wait_tensorcnt(0);
    __syncthreads();
    if (tid < 32 && j + 1 < 16) {
      tdm_load_2d((unsigned)(size_t)&Ba[(j + 1) & 1][0], Wah + (j + 1) * 16,
                  16, 512, 16, 512, 256);
      tdm_load_2d((unsigned)(size_t)&Bb[(j + 1) & 1][0], Wbh + (j + 1) * 16,
                  16, 512, 16, 512, 256);
    }
    const h16* Bap = &Ba[j & 1][0];
    const h16* Bbp = &Bb[j & 1][0];
#else
    __syncthreads();
    for (int i = tid; i < 512 * 16 / 8; i += 256) {
      int f = i * 8; int r = f >> 4, c = f & 15;
      *(uint4*)&Ba[0][r * 16 + c] = *(const uint4*)(Wah + (size_t)r * 256 + j * 16 + c);
      *(uint4*)&Bb[0][r * 16 + c] = *(const uint4*)(Wbh + (size_t)r * 256 + j * 16 + c);
    }
    __syncthreads();
    const h16* Bap = &Ba[0][0];
    const h16* Bbp = &Bb[0][0];
#endif
    v8f aacc, gacc;
    for (int e = 0; e < 8; ++e) { aacc[e] = 0.f; gacc[e] = 0.f; }
#pragma unroll
    for (int ks = 0; ks < 16; ++ks) {
      v16h af  = load_a_frag(Hs, 512, wave * 16, ks * 32);
      v16h bfa = load_b_frag(Bap, 16, ks * 32, 0);
      v16h bfb = load_b_frag(Bbp, 16, ks * 32, 0);
      aacc = __builtin_amdgcn_wmma_f32_16x16x32_f16(false, af, false, bfa, (short)0, aacc, false, false);
      gacc = __builtin_amdgcn_wmma_f32_16x16x32_f16(false, af, false, bfb, (short)0, gacc, false, false);
    }
    int col = j * 16 + nn;
    float bav = ba[col], bbv = bb[col], wcv = Wc[col];
#pragma unroll
    for (int i = 0; i < 8; ++i) {
      float a = tanhf(aacc[i] + bav);
      float g = 1.f / (1.f + expf(-(gacc[i] + bbv)));
      rp[i] += a * g * wcv;
    }
  }
  // reduce across the 16 lanes holding the same row
  for (int i = 0; i < 8; ++i) {
    float v = rp[i];
    for (int off = 1; off < 16; off <<= 1) v += __shfl_xor(v, off, 32);
    if (nn == 0) {
      int grow = row0 + wave * 16 + i + 8 * hf;
      if (grow < N_ROWS) Araw[grow] = v + bc[0];
    }
  }
}

// ---------------- selection: bitonic helpers ----------------
__device__ inline bool cmp_after(float va, int ia, float vb, int ib) {
  // true if (va,ia) comes AFTER (vb,ib) in descending order (ties: smaller idx first)
  return (va < vb) || (va == vb && ia > ib);
}

// phase 1: per-1024-chunk sort; emit top-64 and (negated) bottom-64 candidates
__global__ __launch_bounds__(256) void k_part_sort(const float* __restrict__ A,
                                                   float* __restrict__ ctv, int* __restrict__ cti,
                                                   float* __restrict__ cbv, int* __restrict__ cbi) {
  __shared__ float v[1024];
  __shared__ int   ix[1024];
  const int tid = threadIdx.x;
  const int base = blockIdx.x * 1024;
  const float NEG_INF = -__builtin_inff();

  for (int pass = 0; pass < 2; ++pass) {
    for (int i = tid; i < 1024; i += 256) {
      int g = base + i;
      bool ok = g < N_ROWS;
      float val = ok ? A[g] : 0.f;
      if (pass) val = -val;
      v[i]  = ok ? val : NEG_INF;
      ix[i] = ok ? g : 0x7fffffff;
    }
    __syncthreads();
    for (int k = 2; k <= 1024; k <<= 1)
      for (int jj = k >> 1; jj > 0; jj >>= 1) {
        for (int i = tid; i < 1024; i += 256) {
          int l = i ^ jj;
          if (l > i) {
            bool dirDesc = ((i & k) == 0);
            bool sw = dirDesc ? cmp_after(v[i], ix[i], v[l], ix[l])
                              : cmp_after(v[l], ix[l], v[i], ix[i]);
            if (sw) {
              float tv = v[i]; v[i] = v[l]; v[l] = tv;
              int   ti = ix[i]; ix[i] = ix[l]; ix[l] = ti;
            }
          }
        }
        __syncthreads();
      }
    if (tid < 64) {
      if (pass == 0) { ctv[blockIdx.x * 64 + tid] = v[tid]; cti[blockIdx.x * 64 + tid] = ix[tid]; }
      else           { cbv[blockIdx.x * 64 + tid] = v[tid]; cbi[blockIdx.x * 64 + tid] = ix[tid]; }
    }
    __syncthreads();
  }
}

// phase 2: sort 6272 candidates (padded to 8192); emit final 64+64 indices + global max
__global__ __launch_bounds__(1024) void k_final_sort(const float* __restrict__ ctv, const int* __restrict__ cti,
                                                     const float* __restrict__ cbv, const int* __restrict__ cbi,
                                                     int* __restrict__ selIdx, float* __restrict__ gmaxp) {
  __shared__ float v[8192];
  __shared__ int   ix[8192];
  const int tid = threadIdx.x;
  const float NEG_INF = -__builtin_inff();

  for (int pass = 0; pass < 2; ++pass) {
    const float* cv = pass ? cbv : ctv;
    const int*   ci = pass ? cbi : cti;
    for (int i = tid; i < 8192; i += 1024) {
      bool ok = i < NCAND;
      v[i]  = ok ? cv[i] : NEG_INF;
      ix[i] = ok ? ci[i] : 0x7fffffff;
    }
    __syncthreads();
    for (int k = 2; k <= 8192; k <<= 1)
      for (int jj = k >> 1; jj > 0; jj >>= 1) {
        for (int i = tid; i < 8192; i += 1024) {
          int l = i ^ jj;
          if (l > i) {
            bool dirDesc = ((i & k) == 0);
            bool sw = dirDesc ? cmp_after(v[i], ix[i], v[l], ix[l])
                              : cmp_after(v[l], ix[l], v[i], ix[i]);
            if (sw) {
              float tv = v[i]; v[i] = v[l]; v[l] = tv;
              int   ti = ix[i]; ix[i] = ix[l]; ix[l] = ti;
            }
          }
        }
        __syncthreads();
      }
    if (tid < 64) selIdx[pass * 64 + tid] = ix[tid];
    if (pass == 0 && tid == 0) *gmaxp = v[0];
    __syncthreads();
  }
}

// ---------------- softmax pieces (deterministic tree reductions) ----------------
__global__ __launch_bounds__(256) void k_expsum(const float* __restrict__ A,
                                                const float* __restrict__ gmaxp,
                                                float* __restrict__ partSum) {
  __shared__ float red[256];
  const int tid = threadIdx.x, base = blockIdx.x * 1024;
  float gm = *gmaxp, s = 0.f;
  for (int i = tid; i < 1024; i += 256) {
    int g = base + i;
    if (g < N_ROWS) s += expf(A[g] - gm);
  }
  red[tid] = s; __syncthreads();
  for (int o = 128; o > 0; o >>= 1) { if (tid < o) red[tid] += red[tid + o]; __syncthreads(); }
  if (tid == 0) partSum[blockIdx.x] = red[0];
}

__global__ __launch_bounds__(256) void k_weighted(const float* __restrict__ A,
                                                  const float* __restrict__ gmaxp,
                                                  const h16* __restrict__ h1h,
                                                  float* __restrict__ partM) {
  __shared__ float wbuf[256];
  const int tid = threadIdx.x, base = blockIdx.x * 1024;
  float gm = *gmaxp, a0 = 0.f, a1 = 0.f;
  for (int ch = 0; ch < 4; ++ch) {
    int rb = base + ch * 256;
    int g = rb + tid;
    wbuf[tid] = (g < N_ROWS) ? expf(A[g] - gm) : 0.f;
    __syncthreads();
    int rmax = N_ROWS - rb; if (rmax > 256) rmax = 256; if (rmax < 0) rmax = 0;
    for (int r = 0; r < rmax; ++r) {
      float w = wbuf[r];
      const h16* hp = h1h + (size_t)(rb + r) * 512;
      a0 += w * (float)hp[tid];
      a1 += w * (float)hp[tid + 256];
    }
    __syncthreads();
  }
  partM[blockIdx.x * 512 + tid]       = a0;
  partM[blockIdx.x * 512 + tid + 256] = a1;
}

// ---------------- head: bag classifier + instance classifier + loss ----------------
__global__ __launch_bounds__(512) void k_head(const float* __restrict__ partSum,
                                              const float* __restrict__ partM,
                                              const h16* __restrict__ h1h,
                                              const int* __restrict__ selIdx,
                                              const float* __restrict__ Winst,
                                              const float* __restrict__ binst,
                                              const float* __restrict__ Wcls,
                                              const float* __restrict__ bcls,
                                              const int* __restrict__ labelp,
                                              float* __restrict__ out) {
  __shared__ float Msh[512];
  __shared__ float lg[2];
  __shared__ float Zsh;
  __shared__ float lossArr[128];
  const int tid = threadIdx.x;
  if (tid == 0) { float z = 0.f; for (int b = 0; b < NBLK1K; ++b) z += partSum[b]; Zsh = z; }
  __syncthreads();
  float z = Zsh;
  { float s = 0.f; for (int b = 0; b < NBLK1K; ++b) s += partM[b * 512 + tid]; Msh[tid] = s / z; }
  __syncthreads();
  if (tid < 2) {
    float s = 0.f;
    for (int k = 0; k < 512; ++k) s += Msh[k] * Wcls[k * 2 + tid];
    lg[tid] = s + bcls[tid];
  }
  __syncthreads();
  if (tid == 0) {
    float l0 = lg[0], l1 = lg[1];
    out[0] = l0; out[1] = l1;
    float m = fmaxf(l0, l1);
    float e0 = expf(l0 - m), e1 = expf(l1 - m), se = e0 + e1;
    out[2] = e0 / se; out[3] = e1 / se;
    out[4] = (l1 > l0) ? 1.f : 0.f;
  }
  if (tid < 128) {
    int idx = selIdx[tid];
    const h16* hp = h1h + (size_t)idx * 512;
    float s0 = 0.f, s1 = 0.f;
    for (int k = 0; k < 512; ++k) {
      float hv = (float)hp[k];
      s0 += hv * Winst[k * 2];
      s1 += hv * Winst[k * 2 + 1];
    }
    s0 += binst[0]; s1 += binst[1];
    int pred = (s1 > s0) ? 1 : 0;
    int label = *labelp;
    int tgt = (tid < 64) ? label : 0;
    float m = fmaxf(s0, s1);
    float lse = m + logf(expf(s0 - m) + expf(s1 - m));
    float lt = ((tgt == 1) ? s1 : s0) - lse;
    lossArr[tid] = -lt;
    out[100006 + tid] = (float)pred;
    out[100134 + tid] = (float)tgt;
  }
  __syncthreads();
  if (tid == 0) { float s = 0.f; for (int i = 0; i < 128; ++i) s += lossArr[i]; out[100005] = s / 128.f; }
}

extern "C" void kernel_launch(void* const* d_in, const int* in_sizes, int n_in,
                              void* d_out, int out_size, void* d_ws, size_t ws_size,
                              hipStream_t stream) {
  const float* h     = (const float*)d_in[0];
  const float* W1    = (const float*)d_in[1];
  const float* b1    = (const float*)d_in[2];
  const float* Wa    = (const float*)d_in[3];
  const float* ba    = (const float*)d_in[4];
  const float* Wb    = (const float*)d_in[5];
  const float* bb    = (const float*)d_in[6];
  const float* Wc    = (const float*)d_in[7];
  const float* bc    = (const float*)d_in[8];
  const float* Winst = (const float*)d_in[9];
  const float* binst = (const float*)d_in[10];
  const float* Wcls  = (const float*)d_in[11];
  const float* bcls  = (const float*)d_in[12];
  const int*   label = (const int*)d_in[13];
  float* out = (float*)d_out;

  char* w = (char*)d_ws;
  auto alloc = [&](size_t bytes) { char* p = w; w += (bytes + 255) & ~(size_t)255; return p; };
  h16*   W1h    = (h16*)alloc((size_t)1024 * 512 * 2);
  h16*   Wah    = (h16*)alloc((size_t)512 * 256 * 2);
  h16*   Wbh    = (h16*)alloc((size_t)512 * 256 * 2);
  h16*   h1h    = (h16*)alloc((size_t)N_ROWS * 512 * 2);
  float* ctv    = (float*)alloc((size_t)NCAND * 4);
  int*   cti    = (int*)alloc((size_t)NCAND * 4);
  float* cbv    = (float*)alloc((size_t)NCAND * 4);
  int*   cbi    = (int*)alloc((size_t)NCAND * 4);
  int*   selIdx = (int*)alloc(128 * 4);
  float* gmaxp  = (float*)alloc(4);
  float* partSum= (float*)alloc((size_t)NBLK1K * 4);
  float* partM  = (float*)alloc((size_t)NBLK1K * 512 * 4);

  float* Araw = out + 5;

  hipLaunchKernelGGL(k_cvt, dim3(2048), dim3(256), 0, stream, W1, Wa, Wb, W1h, Wah, Wbh);
  hipLaunchKernelGGL(k_fc1, dim3(NBLK128), dim3(256), 0, stream, h, W1h, b1, h1h);
  hipLaunchKernelGGL(k_attn, dim3(NBLK128), dim3(256), 0, stream, h1h, Wah, Wbh, ba, bb, Wc, bc, Araw);
  hipLaunchKernelGGL(k_part_sort, dim3(NBLK1K), dim3(256), 0, stream, Araw, ctv, cti, cbv, cbi);
  hipLaunchKernelGGL(k_final_sort, dim3(1), dim3(1024), 0, stream, ctv, cti, cbv, cbi, selIdx, gmaxp);
  hipLaunchKernelGGL(k_expsum, dim3(NBLK1K), dim3(256), 0, stream, Araw, gmaxp, partSum);
  hipLaunchKernelGGL(k_weighted, dim3(NBLK1K), dim3(256), 0, stream, Araw, gmaxp, h1h, partM);
  hipLaunchKernelGGL(k_head, dim3(1), dim3(512), 0, stream, partSum, partM, h1h, selIdx,
                     Winst, binst, Wcls, bcls, label, out);
}
